// TridentBottleneck_28080496181597
// MI455X (gfx1250) — matrix operand order repaired and checked by reading, not compiled, not run on hardware
//
#include <hip/hip_runtime.h>
#include <hip/hip_bf16.h>

typedef __attribute__((ext_vector_type(16))) _Float16 v16h;
typedef __attribute__((ext_vector_type(8)))  _Float16 v8h;
typedef __attribute__((ext_vector_type(8)))  float    v8f;

#define NIMG   6
#define CIN    512
#define CB     256
#define COUT   1024
#define HW     64
#define NPIX   4096
#define DGRP   4
#define KOFF   72
#define MOFF   80
#define KD     2304      // 9 * 256
#define HPAD   66
#define PLANE  4356      // 66*66
#define EPSBN  1e-5f

// ---------------- WMMA fragment helpers (CDNA5 16x16x32 f16 layouts) ----------------

__device__ __forceinline__ v8f wmma32(v16h a, v16h b, v8f c) {
  return __builtin_amdgcn_wmma_f32_16x16x32_f16(false, a, false, b, (short)0, c,
                                                false, false);
}

// A fragment: row-major W[M][ld], tile at (m0,k0). lane -> M=m0+(lane&15);
// halves 0..7 = K kb..kb+7, halves 8..15 = K kb+16..kb+23, kb = 8*(lane>>4).
__device__ __forceinline__ v16h frag_a(const _Float16* W, int ld, int m0, int k0, int lane) {
  const _Float16* p = W + (size_t)(m0 + (lane & 15)) * ld + (k0 + ((lane >> 4) << 3));
  v16h r;
  __builtin_memcpy(&r, p, 16);
  __builtin_memcpy(reinterpret_cast<_Float16*>(&r) + 8, p + 16, 16);
  return r;
}

// B fragment: lane -> N column at p, K rows kb..kb+7 and kb+16..kb+23 (stride ld), kb = 8*(lane>>4) folded into p.
__device__ __forceinline__ v16h frag_b(const _Float16* p, int ld) {
  v16h r;
  #pragma unroll
  for (int s = 0; s < 8; ++s) {
    r[s] = p[(size_t)s * ld];
    r[s + 8] = p[(size_t)(s + 16) * ld];
  }
  return r;
}

// ---------------- prep kernels ----------------

__global__ void k_convert_x(const float* __restrict__ x, _Float16* __restrict__ xf, int n) {
  int i = blockIdx.x * 256 + threadIdx.x;
  if (i < n) xf[i] = (_Float16)x[i];
}

__global__ void k_zero32(unsigned int* __restrict__ p, int n32) {
  int i = blockIdx.x * 256 + threadIdx.x;
  if (i < n32) p[i] = 0u;
}

// fold BN scale into 1x1-conv weights: Wf[o][i] = w[o][i]*s(o), T[o] = b - m*s
__global__ void k_prep_lin(const float* __restrict__ w, const float* __restrict__ g,
                           const float* __restrict__ b, const float* __restrict__ m,
                           const float* __restrict__ v, _Float16* __restrict__ Wf,
                           float* __restrict__ T, int O, int I) {
  int idx = blockIdx.x * 256 + threadIdx.x;
  if (idx >= O * I) return;
  int o = idx / I, i = idx - o * I;
  float s = g[o] * rsqrtf(v[o] + EPSBN);
  Wf[idx] = (_Float16)(w[idx] * s);
  if (i == 0) T[o] = b[o] - m[o] * s;
}

// deform-conv weights: W2f[o][kk*256+c] = w2[o][c][kk] * s2(o)
__global__ void k_prep_w2(const float* __restrict__ w2, const float* __restrict__ g,
                          const float* __restrict__ b, const float* __restrict__ m,
                          const float* __restrict__ v, _Float16* __restrict__ W2f,
                          float* __restrict__ T2) {
  int idx = blockIdx.x * 256 + threadIdx.x;
  if (idx >= CB * KD) return;
  int o = idx / KD, r = idx - o * KD;
  int kk = r >> 8, c = r & 255;
  float s = g[o] * rsqrtf(v[o] + EPSBN);
  W2f[idx] = (_Float16)(w2[(size_t)(o * CB + c) * 9 + kk] * s);
  if (r == 0) T2[o] = b[o] - m[o] * s;
}

// offset-conv weights, padded to 80 rows: Woff[mm][kk*256+c]
__global__ void k_prep_woff(const float* __restrict__ w_off, const float* __restrict__ b_off,
                            _Float16* __restrict__ Woff, float* __restrict__ BOFF) {
  int idx = blockIdx.x * 256 + threadIdx.x;
  if (idx >= MOFF * KD) return;
  int mm = idx / KD, r = idx - mm * KD;
  int kk = r >> 8, c = r & 255;
  _Float16 val = (_Float16)0.0f;
  if (mm < KOFF) val = (_Float16)w_off[(size_t)(mm * CB + c) * 9 + kk];
  Woff[idx] = val;
  if (mm < KOFF && r == 0) BOFF[mm] = b_off[mm];
}

// ---------------- GEMM kernels (32x64 wave tile, 64x256 block tile) ----------------

// H1 = relu(W1f * Xf + T1); stored twice:
//  - H1p: f16 channel-major with 1-px zero halo  [img][c][66][66]  (for offset conv)
//  - H1c: f16 channel-last                        [img][pix][256]   (for bilinear gather)
__global__ __launch_bounds__(256) void k_gemm_h1(const _Float16* __restrict__ W1f,
                                                 const _Float16* __restrict__ Xf,
                                                 const float* __restrict__ T1,
                                                 _Float16* __restrict__ H1p,
                                                 _Float16* __restrict__ H1c) {
  int lane = threadIdx.x & 31, wave = threadIdx.x >> 5;
  int img = blockIdx.z;
  int m0 = blockIdx.y * 64 + (wave >> 2) * 32;
  int n0 = blockIdx.x * 256 + (wave & 3) * 64;
  const _Float16* Bimg = Xf + (size_t)img * CIN * NPIX;
  v8f acc[2][4] = {};
  for (int k = 0; k < CIN; k += 32) {
    v16h a0 = frag_a(W1f, CIN, m0, k, lane);
    v16h a1 = frag_a(W1f, CIN, m0 + 16, k, lane);
    const _Float16* brow = Bimg + (size_t)(k + ((lane >> 4) << 3)) * NPIX + n0 + (lane & 15);
    #pragma unroll
    for (int j = 0; j < 4; ++j) {
      v16h b = frag_b(brow + j * 16, NPIX);
      acc[0][j] = wmma32(a0, b, acc[0][j]);
      acc[1][j] = wmma32(a1, b, acc[1][j]);
    }
  }
  _Float16* Himg = H1p + (size_t)img * CB * PLANE;
  _Float16* Hc   = H1c + (size_t)img * NPIX * CB;
  #pragma unroll
  for (int i = 0; i < 2; ++i)
    #pragma unroll
    for (int j = 0; j < 4; ++j) {
      int n = n0 + j * 16 + (lane & 15);
      int pidx = ((n >> 6) + 1) * HPAD + (n & 63) + 1;
      int mrow = m0 + i * 16 + ((lane >> 4) << 3);
      v8h hv;
      #pragma unroll
      for (int r = 0; r < 8; ++r) {
        int mm = mrow + r;
        float val = acc[i][j][r] + T1[mm];
        val = val > 0.0f ? val : 0.0f;
        _Float16 hval = (_Float16)val;
        Himg[(size_t)mm * PLANE + pidx] = hval;
        hv[r] = hval;
      }
      *reinterpret_cast<v8h*>(Hc + (size_t)n * CB + mrow) = hv;
    }
}

// OFF = Woff (*) H1 + b_off  (implicit-GEMM 3x3 conv over halo'd H1p), f32 out
__global__ __launch_bounds__(256) void k_gemm_off(const _Float16* __restrict__ Woff,
                                                  const _Float16* __restrict__ H1p,
                                                  const float* __restrict__ BOFF,
                                                  float* __restrict__ OFF) {
  int lane = threadIdx.x & 31, wave = threadIdx.x >> 5;
  int img = blockIdx.z;
  int m0 = blockIdx.y * 16;
  int n0 = blockIdx.x * 512 + wave * 64;      // 64 pixels = one image row chunk
  int y = n0 >> 6;                            // n0 multiple of 64 -> single row
  const _Float16* Himg = H1p + (size_t)img * CB * PLANE;
  v8f acc[4] = {};
  for (int k = 0; k < KD; k += 32) {
    v16h a = frag_a(Woff, KD, m0, k, lane);
    int Kidx = k + ((lane >> 4) << 3);
    int kk = Kidx >> 8;
    int c = Kidx & 255;
    int dy = kk / 3 - 1;
    int dx = kk - (kk / 3) * 3 - 1;
    const _Float16* brow = Himg + (size_t)c * PLANE + (size_t)(y + dy + 1) * HPAD + (dx + 1) + (lane & 15);
    #pragma unroll
    for (int j = 0; j < 4; ++j) {
      v16h b = frag_b(brow + j * 16, PLANE);
      acc[j] = wmma32(a, b, acc[j]);
    }
  }
  float* Oimg = OFF + (size_t)img * KOFF * NPIX;
  #pragma unroll
  for (int j = 0; j < 4; ++j) {
    int n = n0 + j * 16 + (lane & 15);
    int mrow = m0 + ((lane >> 4) << 3);
    #pragma unroll
    for (int r = 0; r < 8; ++r) {
      int mm = mrow + r;
      if (mm < KOFF) Oimg[(size_t)mm * NPIX + n] = acc[j][r] + BOFF[mm];
    }
  }
}

// bilinear im2col from channel-last H1c: COL[kk*256+c][pix] f16.
// One thread per (g,kk,pixel); 64 channels via contiguous v8h (b128) corner loads.
__global__ __launch_bounds__(256) void k_sample(const _Float16* __restrict__ H1c,
                                                const float* __restrict__ OFF,
                                                _Float16* __restrict__ COL, int img) {
  int pix = blockIdx.x * 256 + threadIdx.x;
  int gk = blockIdx.y;                 // 0..35
  int g = gk / 9, kk = gk - g * 9;
  int y = pix >> 6, x = pix & 63;
  int dy = kk / 3 - 1;
  int dx = kk - (kk / 3) * 3 - 1;
  const float* offp = OFF + ((size_t)img * KOFF + gk * 2) * NPIX;
  float py = (float)(y + dy) + offp[pix];
  float px = (float)(x + dx) + offp[NPIX + pix];
  float fy0 = floorf(py), fx0 = floorf(px);
  float wy = py - fy0, wx = px - fx0;
  int y0 = (int)fy0, x0 = (int)fx0;
  int y1 = y0 + 1, x1 = x0 + 1;
  float vy0 = (y0 >= 0 && y0 < HW) ? 1.0f : 0.0f;
  float vy1 = (y1 >= 0 && y1 < HW) ? 1.0f : 0.0f;
  float vx0 = (x0 >= 0 && x0 < HW) ? 1.0f : 0.0f;
  float vx1 = (x1 >= 0 && x1 < HW) ? 1.0f : 0.0f;
  int y0c = min(max(y0, 0), HW - 1), y1c = min(max(y1, 0), HW - 1);
  int x0c = min(max(x0, 0), HW - 1), x1c = min(max(x1, 0), HW - 1);
  float w00 = (1.0f - wy) * (1.0f - wx) * vy0 * vx0;
  float w01 = (1.0f - wy) * wx * vy0 * vx1;
  float w10 = wy * (1.0f - wx) * vy1 * vx0;
  float w11 = wy * wx * vy1 * vx1;
  const _Float16* base = H1c + (size_t)img * NPIX * CB + g * 64;
  const _Float16* p00 = base + (size_t)((y0c << 6) | x0c) * CB;
  const _Float16* p01 = base + (size_t)((y0c << 6) | x1c) * CB;
  const _Float16* p10 = base + (size_t)((y1c << 6) | x0c) * CB;
  const _Float16* p11 = base + (size_t)((y1c << 6) | x1c) * CB;
  _Float16* colp = COL + (size_t)(kk * CB + g * 64) * NPIX + pix;
  #pragma unroll
  for (int c8 = 0; c8 < 8; ++c8) {
    v8h q00 = *reinterpret_cast<const v8h*>(p00 + c8 * 8);
    v8h q01 = *reinterpret_cast<const v8h*>(p01 + c8 * 8);
    v8h q10 = *reinterpret_cast<const v8h*>(p10 + c8 * 8);
    v8h q11 = *reinterpret_cast<const v8h*>(p11 + c8 * 8);
    #pragma unroll
    for (int e = 0; e < 8; ++e) {
      float s = w00 * (float)q00[e] + w01 * (float)q01[e] +
                w10 * (float)q10[e] + w11 * (float)q11[e];
      colp[(size_t)(c8 * 8 + e) * NPIX] = (_Float16)s;
    }
  }
}

// H2 = relu(W2f * COL + T2), f16 [img][c][pix]
__global__ __launch_bounds__(256) void k_gemm_h2(const _Float16* __restrict__ W2f,
                                                 const _Float16* __restrict__ COL,
                                                 const float* __restrict__ T2,
                                                 _Float16* __restrict__ H2, int img) {
  int lane = threadIdx.x & 31, wave = threadIdx.x >> 5;
  int m0 = blockIdx.y * 64 + (wave >> 2) * 32;
  int n0 = blockIdx.x * 256 + (wave & 3) * 64;
  v8f acc[2][4] = {};
  for (int k = 0; k < KD; k += 32) {
    v16h a0 = frag_a(W2f, KD, m0, k, lane);
    v16h a1 = frag_a(W2f, KD, m0 + 16, k, lane);
    const _Float16* brow = COL + (size_t)(k + ((lane >> 4) << 3)) * NPIX + n0 + (lane & 15);
    #pragma unroll
    for (int j = 0; j < 4; ++j) {
      v16h b = frag_b(brow + j * 16, NPIX);
      acc[0][j] = wmma32(a0, b, acc[0][j]);
      acc[1][j] = wmma32(a1, b, acc[1][j]);
    }
  }
  _Float16* Himg = H2 + (size_t)img * CB * NPIX;
  #pragma unroll
  for (int i = 0; i < 2; ++i)
    #pragma unroll
    for (int j = 0; j < 4; ++j) {
      int n = n0 + j * 16 + (lane & 15);
      int mrow = m0 + i * 16 + ((lane >> 4) << 3);
      #pragma unroll
      for (int r = 0; r < 8; ++r) {
        int mm = mrow + r;
        float val = acc[i][j][r] + T2[mm];
        val = val > 0.0f ? val : 0.0f;
        Himg[(size_t)mm * NPIX + n] = (_Float16)val;
      }
    }
}

// out = relu(W3f*H2 + Wdf*Xf + T3 + TD), f32 to d_out [6][1024][4096]
__global__ __launch_bounds__(256) void k_gemm_out(const _Float16* __restrict__ W3f,
                                                  const _Float16* __restrict__ Wdf,
                                                  const _Float16* __restrict__ H2,
                                                  const _Float16* __restrict__ Xf,
                                                  const float* __restrict__ T3,
                                                  const float* __restrict__ TD,
                                                  float* __restrict__ out) {
  int lane = threadIdx.x & 31, wave = threadIdx.x >> 5;
  int img = blockIdx.z;
  int m0 = blockIdx.y * 64 + (wave >> 2) * 32;
  int n0 = blockIdx.x * 256 + (wave & 3) * 64;
  v8f acc[2][4] = {};
  const _Float16* B1 = H2 + (size_t)img * CB * NPIX;
  for (int k = 0; k < CB; k += 32) {
    v16h a0 = frag_a(W3f, CB, m0, k, lane);
    v16h a1 = frag_a(W3f, CB, m0 + 16, k, lane);
    const _Float16* brow = B1 + (size_t)(k + ((lane >> 4) << 3)) * NPIX + n0 + (lane & 15);
    #pragma unroll
    for (int j = 0; j < 4; ++j) {
      v16h b = frag_b(brow + j * 16, NPIX);
      acc[0][j] = wmma32(a0, b, acc[0][j]);
      acc[1][j] = wmma32(a1, b, acc[1][j]);
    }
  }
  const _Float16* B2 = Xf + (size_t)img * CIN * NPIX;
  for (int k = 0; k < CIN; k += 32) {
    v16h a0 = frag_a(Wdf, CIN, m0, k, lane);
    v16h a1 = frag_a(Wdf, CIN, m0 + 16, k, lane);
    const _Float16* brow = B2 + (size_t)(k + ((lane >> 4) << 3)) * NPIX + n0 + (lane & 15);
    #pragma unroll
    for (int j = 0; j < 4; ++j) {
      v16h b = frag_b(brow + j * 16, NPIX);
      acc[0][j] = wmma32(a0, b, acc[0][j]);
      acc[1][j] = wmma32(a1, b, acc[1][j]);
    }
  }
  float* Oimg = out + (size_t)img * COUT * NPIX;
  #pragma unroll
  for (int i = 0; i < 2; ++i)
    #pragma unroll
    for (int j = 0; j < 4; ++j) {
      int n = n0 + j * 16 + (lane & 15);
      int mrow = m0 + i * 16 + ((lane >> 4) << 3);
      #pragma unroll
      for (int r = 0; r < 8; ++r) {
        int mm = mrow + r;
        float val = acc[i][j][r] + T3[mm] + TD[mm];
        val = val > 0.0f ? val : 0.0f;
        Oimg[(size_t)mm * NPIX + n] = val;
      }
    }
}

// ---------------- launcher ----------------

extern "C" void kernel_launch(void* const* d_in, const int* in_sizes, int n_in,
                              void* d_out, int out_size, void* d_ws, size_t ws_size,
                              hipStream_t stream) {
  const float* x     = (const float*)d_in[0];
  const float* w1    = (const float*)d_in[1];
  const float* g1    = (const float*)d_in[2];
  const float* b1    = (const float*)d_in[3];
  const float* m1    = (const float*)d_in[4];
  const float* v1    = (const float*)d_in[5];
  const float* w_off = (const float*)d_in[6];
  const float* b_off = (const float*)d_in[7];
  const float* w2    = (const float*)d_in[8];
  const float* g2    = (const float*)d_in[9];
  const float* b2    = (const float*)d_in[10];
  const float* m2    = (const float*)d_in[11];
  const float* v2    = (const float*)d_in[12];
  const float* w3    = (const float*)d_in[13];
  const float* g3    = (const float*)d_in[14];
  const float* b3    = (const float*)d_in[15];
  const float* m3    = (const float*)d_in[16];
  const float* v3    = (const float*)d_in[17];
  const float* wd    = (const float*)d_in[18];
  const float* gd    = (const float*)d_in[19];
  const float* bd    = (const float*)d_in[20];
  const float* md    = (const float*)d_in[21];
  const float* vd    = (const float*)d_in[22];
  float* out = (float*)d_out;

  size_t off = 0;
  auto carve = [&](size_t bytes) -> void* {
    void* p = (char*)d_ws + off;
    off += (bytes + 255) & ~(size_t)255;
    return p;
  };
  _Float16* Xf   = (_Float16*)carve(sizeof(_Float16) * (size_t)NIMG * CIN * NPIX);
  _Float16* W1f  = (_Float16*)carve(sizeof(_Float16) * (size_t)CB * CIN);
  _Float16* Wdf  = (_Float16*)carve(sizeof(_Float16) * (size_t)COUT * CIN);
  _Float16* W3f  = (_Float16*)carve(sizeof(_Float16) * (size_t)COUT * CB);
  _Float16* W2f  = (_Float16*)carve(sizeof(_Float16) * (size_t)CB * KD);
  _Float16* Woff = (_Float16*)carve(sizeof(_Float16) * (size_t)MOFF * KD);
  _Float16* H1p  = (_Float16*)carve(sizeof(_Float16) * (size_t)NIMG * CB * PLANE);
  _Float16* H1c  = (_Float16*)carve(sizeof(_Float16) * (size_t)NIMG * NPIX * CB);
  _Float16* H2   = (_Float16*)carve(sizeof(_Float16) * (size_t)NIMG * CB * NPIX);
  _Float16* COL  = (_Float16*)carve(sizeof(_Float16) * (size_t)KD * NPIX);
  float* OFFb = (float*)carve(sizeof(float) * (size_t)NIMG * KOFF * NPIX);
  float* T1   = (float*)carve(sizeof(float) * CB);
  float* T2   = (float*)carve(sizeof(float) * CB);
  float* T3   = (float*)carve(sizeof(float) * COUT);
  float* TD   = (float*)carve(sizeof(float) * COUT);
  float* BOFF = (float*)carve(sizeof(float) * KOFF);

  // prep
  {
    int n = NIMG * CIN * NPIX;
    k_convert_x<<<(n + 255) / 256, 256, 0, stream>>>(x, Xf, n);
  }
  {
    int n32 = NIMG * CB * PLANE / 2;
    k_zero32<<<(n32 + 255) / 256, 256, 0, stream>>>((unsigned int*)H1p, n32);
  }
  k_prep_lin<<<(CB * CIN + 255) / 256, 256, 0, stream>>>(w1, g1, b1, m1, v1, W1f, T1, CB, CIN);
  k_prep_lin<<<(COUT * CIN + 255) / 256, 256, 0, stream>>>(wd, gd, bd, md, vd, Wdf, TD, COUT, CIN);
  k_prep_lin<<<(COUT * CB + 255) / 256, 256, 0, stream>>>(w3, g3, b3, m3, v3, W3f, T3, COUT, CB);
  k_prep_w2<<<(CB * KD + 255) / 256, 256, 0, stream>>>(w2, g2, b2, m2, v2, W2f, T2);
  k_prep_woff<<<(MOFF * KD + 255) / 256, 256, 0, stream>>>(w_off, b_off, Woff, BOFF);

  // stage 1: conv1 + BN + relu (dual layout: halo'd channel-major + channel-last)
  k_gemm_h1<<<dim3(NPIX / 256, CB / 64, NIMG), 256, 0, stream>>>(W1f, Xf, T1, H1p, H1c);
  // stage 2: offset conv (3x3 implicit GEMM) + bias
  k_gemm_off<<<dim3(NPIX / 512, MOFF / 16, NIMG), 256, 0, stream>>>(Woff, H1p, BOFF, OFFb);
  // stage 3+4: per-image bilinear im2col then deform GEMM (+BN+relu)
  for (int img = 0; img < NIMG; ++img) {
    k_sample<<<dim3(NPIX / 256, DGRP * 9), 256, 0, stream>>>(H1c, OFFb, COL, img);
    k_gemm_h2<<<dim3(NPIX / 256, CB / 64), 256, 0, stream>>>(W2f, COL, T2, H2, img);
  }
  // stage 5: conv3 + downsample fused, + biases, relu
  k_gemm_out<<<dim3(NPIX / 256, COUT / 64, NIMG), 256, 0, stream>>>(W3f, Wdf, H2, Xf, T3, TD, out);
}
